// Self_Attention_Layer_54443005444381
// MI455X (gfx1250) — matrix operand order repaired and checked by reading, not compiled, not run on hardware
//
#include <hip/hip_runtime.h>

typedef __bf16 bf16_t;
typedef __attribute__((ext_vector_type(16))) __bf16 v16bf;
typedef __attribute__((ext_vector_type(8)))  __bf16 v8bf;
typedef __attribute__((ext_vector_type(8)))  float  v8f;

#define N_PTS 32768
#define NBAT  2
#define CCH   64
#define KNN   16
#define WAVES 4            // waves per block (128 threads)
#define PTS_PER_WAVE 16

// ---------------- operand fragment loaders (CDNA5 WMMA bf16 layouts) ----------
// A 16x32 bf16: lane holds row (l&15); elems 0..15 = k[32s + 16*(l>=16) + 0..15]
//               (contiguous, matching fragB)
__device__ __forceinline__ v16bf fragA(const bf16_t* row, int kbase, int colbase) {
  v8bf lo = *(const v8bf*)(row + kbase + 2 * colbase);
  v8bf hi = *(const v8bf*)(row + kbase + 2 * colbase + 8);
  v16bf r;
#pragma unroll
  for (int i = 0; i < 8; ++i) { r[i] = lo[i]; r[8 + i] = hi[i]; }
  return r;
}
// B 32x16 bf16: lane holds column (l&15); elems 0..15 = k[32s + 16*(l>=16) + 0..15]
// (row striped across lanes within a VGPR; contiguous k order chosen here, same as fragA)
__device__ __forceinline__ v16bf fragB(const bf16_t* row, int kbase) {
  v8bf lo = *(const v8bf*)(row + kbase);
  v8bf hi = *(const v8bf*)(row + kbase + 8);
  v16bf r;
#pragma unroll
  for (int i = 0; i < 8; ++i) { r[i] = lo[i]; r[8 + i] = hi[i]; }
  return r;
}

// Project one 16x64 A-tile through Wq/Wk/Wv (all bf16, row-major [out][in]),
// bias-add, store rows into LDS tiles (row stride 72 bf16, conflict-free).
__device__ __forceinline__ void project_tile(
    v16bf a0, v16bf a1,
    const bf16_t* __restrict__ wq, const bf16_t* __restrict__ wk,
    const bf16_t* __restrict__ wv, const float* sB /*[3*64]*/,
    bf16_t* Qd, bf16_t* Kd, bf16_t* Vd, int l15, int lh) {
  const int rbase = lh * 8;
#pragma unroll
  for (int t = 0; t < 4; ++t) {
    const int n = t * 16 + l15;                 // output channel = B column
    {
      v16bf b0 = fragB(wq + n * 64, lh * 16);
      v16bf b1 = fragB(wq + n * 64, 32 + lh * 16);
      v8f acc = {0.f,0.f,0.f,0.f,0.f,0.f,0.f,0.f};
      acc = __builtin_amdgcn_wmma_f32_16x16x32_bf16(false, a0, false, b0, (short)0, acc, false, false);
      acc = __builtin_amdgcn_wmma_f32_16x16x32_bf16(false, a1, false, b1, (short)0, acc, false, false);
      float bias = sB[n];
#pragma unroll
      for (int i = 0; i < 8; ++i) Qd[(rbase + i) * 72 + n] = (bf16_t)(acc[i] + bias);
    }
    {
      v16bf b0 = fragB(wk + n * 64, lh * 16);
      v16bf b1 = fragB(wk + n * 64, 32 + lh * 16);
      v8f acc = {0.f,0.f,0.f,0.f,0.f,0.f,0.f,0.f};
      acc = __builtin_amdgcn_wmma_f32_16x16x32_bf16(false, a0, false, b0, (short)0, acc, false, false);
      acc = __builtin_amdgcn_wmma_f32_16x16x32_bf16(false, a1, false, b1, (short)0, acc, false, false);
      float bias = sB[64 + n];
#pragma unroll
      for (int i = 0; i < 8; ++i) Kd[(rbase + i) * 72 + n] = (bf16_t)(acc[i] + bias);
    }
    {
      v16bf b0 = fragB(wv + n * 64, lh * 16);
      v16bf b1 = fragB(wv + n * 64, 32 + lh * 16);
      v8f acc = {0.f,0.f,0.f,0.f,0.f,0.f,0.f,0.f};
      acc = __builtin_amdgcn_wmma_f32_16x16x32_bf16(false, a0, false, b0, (short)0, acc, false, false);
      acc = __builtin_amdgcn_wmma_f32_16x16x32_bf16(false, a1, false, b1, (short)0, acc, false, false);
      float bias = sB[128 + n];
#pragma unroll
      for (int i = 0; i < 8; ++i) Vd[(rbase + i) * 72 + n] = (bf16_t)(acc[i] + bias);
    }
  }
}

// ---------------- prep kernels ------------------------------------------------
__global__ void k_prep_w(const float* __restrict__ Wq, const float* __restrict__ Wk,
                         const float* __restrict__ Wv, bf16_t* __restrict__ wbf) {
  int i = blockIdx.x * 256 + threadIdx.x;     // 12288 total
  const float* s = (i < 4096) ? Wq : (i < 8192) ? Wk : Wv;
  wbf[i] = (bf16_t)s[i & 4095];
}

// feature [B,C,N] f32 -> featT [B*N, 64] bf16 (tiled transpose via LDS)
__global__ __launch_bounds__(256) void k_transpose(const float* __restrict__ f,
                                                   bf16_t* __restrict__ featT) {
  __shared__ float tile[64][65];
  int b = blockIdx.x >> 9;                    // 512 tiles of 64 points per batch
  int nbase = (blockIdx.x & 511) * 64;
  int tn = threadIdx.x & 63, c4 = threadIdx.x >> 6;
#pragma unroll 4
  for (int i = 0; i < 16; ++i) {
    int c = c4 + i * 4;
    tile[c][tn] = f[(b * 64 + c) * N_PTS + nbase + tn];
  }
  __syncthreads();
  int cc = threadIdx.x & 63, n4 = threadIdx.x >> 6;
#pragma unroll 4
  for (int i = 0; i < 16; ++i) {
    int n = n4 + i * 4;
    featT[(b * N_PTS + nbase + n) * 64 + cc] = (bf16_t)tile[cc][n];
  }
}

// ---------------- main attention kernel --------------------------------------
__global__ __launch_bounds__(128) void k_main(
    const bf16_t* __restrict__ featT, const long long* __restrict__ knn,
    const bf16_t* __restrict__ wbf, const float* __restrict__ bq,
    const float* __restrict__ bk, const float* __restrict__ bv,
    float* __restrict__ xbuf) {
  __shared__ bf16_t sQc[WAVES][16][72], sKc[WAVES][16][72], sVc[WAVES][16][72];
  __shared__ bf16_t sQn[WAVES][16][72], sKn[WAVES][16][72], sVn[WAVES][16][72];
  __shared__ float  sSim[WAVES][17][20];
  __shared__ float  sWgt[WAVES][20];
  __shared__ float  sBias[3 * 64];

  const int tid = threadIdx.x;
  if (tid < 64) { sBias[tid] = bq[tid]; sBias[64 + tid] = bk[tid]; sBias[128 + tid] = bv[tid]; }
  __syncthreads();

  const int wid = tid >> 5, lane = tid & 31;
  const int l15 = lane & 15, lh = lane >> 4;
  const int colbase = lh * 8, rbase = lh * 8;
  const bf16_t* wq = wbf, *wk = wbf + 4096, *wv = wbf + 8192;

  const int Pw = blockIdx.x * (WAVES * PTS_PER_WAVE) + wid * PTS_PER_WAVE;
  const int b = Pw >> 15;                    // 64-point chunks never straddle batches
  const int browbase = b << 15;

  // ---- center tile: 16 centers of this wave's 16 points = one perfect A-tile
  {
    const bf16_t* crow = featT + (Pw + l15) * 64;
    v16bf a0 = fragA(crow, 0, colbase);
    v16bf a1 = fragA(crow, 32, colbase);
    project_tile(a0, a1, wq, wk, wv, sBias, &sQc[wid][0][0], &sKc[wid][0][0],
                 &sVc[wid][0][0], l15, lh);
  }

  // ---- per-point neighbor tiles
  for (int p = 0; p < PTS_PER_WAVE; ++p) {
    const int gp = Pw + p;
    const int idx = (int)knn[gp * KNN + l15];          // lane row = neighbor l15
    const bf16_t* nrow = featT + (browbase + idx) * 64;
    v16bf a0 = fragA(nrow, 0, colbase);
    v16bf a1 = fragA(nrow, 32, colbase);
    project_tile(a0, a1, wq, wk, wv, sBias, &sQn[wid][0][0], &sKn[wid][0][0],
                 &sVn[wid][0][0], l15, lh);

    // sim[0:16,0:16] = Qn @ Kn^T via WMMA (B column j = Kn row j)
    {
      v16bf qa0 = fragA(&sQn[wid][l15][0], 0, colbase);
      v16bf qa1 = fragA(&sQn[wid][l15][0], 32, colbase);
      v16bf kb0 = fragB(&sKn[wid][l15][0], lh * 16);
      v16bf kb1 = fragB(&sKn[wid][l15][0], 32 + lh * 16);
      v8f acc = {0.f,0.f,0.f,0.f,0.f,0.f,0.f,0.f};
      acc = __builtin_amdgcn_wmma_f32_16x16x32_bf16(false, qa0, false, kb0, (short)0, acc, false, false);
      acc = __builtin_amdgcn_wmma_f32_16x16x32_bf16(false, qa1, false, kb1, (short)0, acc, false, false);
#pragma unroll
      for (int i = 0; i < 8; ++i) sSim[wid][rbase + i][l15] = acc[i];
    }

    // sim border (center row/col) on VALU: 33 dot64's across the wave
    const bf16_t* Qc = &sQc[wid][p][0];
    const bf16_t* Kc = &sKc[wid][p][0];
    if (lane < 17) {                                   // sim[16][j]
      const bf16_t* kr = (lane < 16) ? &sKn[wid][lane][0] : Kc;
      float d = 0.f;
#pragma unroll
      for (int c8 = 0; c8 < 8; ++c8) {
        v8bf q = *(const v8bf*)(Qc + c8 * 8);
        v8bf k = *(const v8bf*)(kr + c8 * 8);
#pragma unroll
        for (int e = 0; e < 8; ++e) d += (float)q[e] * (float)k[e];
      }
      sSim[wid][16][lane] = d;
    }
    if (lane < 16) {                                   // sim[i][16]
      const bf16_t* qr = &sQn[wid][lane][0];
      float d = 0.f;
#pragma unroll
      for (int c8 = 0; c8 < 8; ++c8) {
        v8bf q = *(const v8bf*)(qr + c8 * 8);
        v8bf k = *(const v8bf*)(Kc + c8 * 8);
#pragma unroll
        for (int e = 0; e < 8; ++e) d += (float)q[e] * (float)k[e];
      }
      sSim[wid][lane][16] = d;
    }

    // diagonal softmax weights: w_i = exp(sim[i][i]-m_i) / sum_j exp(sim[i][j]-m_i)
    if (lane < 17) {
      float m = -1e30f;
#pragma unroll
      for (int j = 0; j < 17; ++j) m = fmaxf(m, sSim[wid][lane][j]);
      float s = 0.f;
#pragma unroll
      for (int j = 0; j < 17; ++j) s += __expf(sSim[wid][lane][j] - m);
      sWgt[wid][lane] = __expf(sSim[wid][lane][lane] - m) / s;
    }

    // x[c] = sum_i w_i * V_i[c]; lanes cover c = lane, lane+32
    float x0 = 0.f, x1 = 0.f;
#pragma unroll
    for (int i = 0; i < 16; ++i) {
      float wv_ = sWgt[wid][i];
      x0 += wv_ * (float)sVn[wid][i][lane];
      x1 += wv_ * (float)sVn[wid][i][lane + 32];
    }
    {
      float wv_ = sWgt[wid][16];
      x0 += wv_ * (float)sVc[wid][p][lane];
      x1 += wv_ * (float)sVc[wid][p][lane + 32];
    }
    xbuf[gp * 64 + lane] = x0;
    xbuf[gp * 64 + lane + 32] = x1;
  }
}

// ---------------- batch-norm stats + apply -----------------------------------
__global__ __launch_bounds__(256) void k_stats(const float* __restrict__ xbuf,
                                               float* __restrict__ gsum,
                                               float* __restrict__ gsq) {
  __shared__ float ls[64], lq[64];
  int tid = threadIdx.x;
  if (tid < 64) { ls[tid] = 0.f; lq[tid] = 0.f; }
  __syncthreads();
  int c = tid & 63;
  float s = 0.f, q = 0.f;
  for (int r = blockIdx.x * 4 + (tid >> 6); r < NBAT * N_PTS; r += gridDim.x * 4) {
    float x = xbuf[r * 64 + c];
    s += x; q += x * x;
  }
  atomicAdd(&ls[c], s);
  atomicAdd(&lq[c], q);
  __syncthreads();
  if (tid < 64) { atomicAdd(&gsum[tid], ls[tid]); atomicAdd(&gsq[tid], lq[tid]); }
}

__global__ __launch_bounds__(256) void k_apply(
    const float* __restrict__ f, const float* __restrict__ xbuf,
    const float* __restrict__ gsum, const float* __restrict__ gsq,
    const float* __restrict__ gamma, const float* __restrict__ beta,
    float* __restrict__ out) {
  __shared__ float tile[64][65];
  int b = blockIdx.x >> 9;
  int nbase = (blockIdx.x & 511) * 64;
  int c = threadIdx.x & 63, n4 = threadIdx.x >> 6;
  const float inv = 1.0f / (float)(NBAT * N_PTS);
  float mean = gsum[c] * inv;
  float var = gsq[c] * inv - mean * mean;
  float rstd = rsqrtf(var + 1e-5f);
  float g = gamma[c] * rstd, be = beta[c];
#pragma unroll 4
  for (int i = 0; i < 16; ++i) {
    int n = n4 + i * 4;
    float x = xbuf[(b * N_PTS + nbase + n) * 64 + c];
    float v = (x - mean) * g + be;
    tile[c][n] = v > 0.f ? v : 0.f;
  }
  __syncthreads();
  int tn = threadIdx.x & 63, c4 = threadIdx.x >> 6;
#pragma unroll 4
  for (int i = 0; i < 16; ++i) {
    int cc = c4 + i * 4;
    int idx = (b * 64 + cc) * N_PTS + nbase + tn;
    out[idx] = f[idx] + tile[cc][tn];
  }
}

// ---------------- host orchestration -----------------------------------------
extern "C" void kernel_launch(void* const* d_in, const int* in_sizes, int n_in,
                              void* d_out, int out_size, void* d_ws, size_t ws_size,
                              hipStream_t stream) {
  (void)in_sizes; (void)n_in; (void)out_size; (void)ws_size;
  const float*     feature = (const float*)d_in[0];
  const long long* knn     = (const long long*)d_in[1];
  const float*     Wq      = (const float*)d_in[2];
  const float*     bq      = (const float*)d_in[3];
  const float*     Wk      = (const float*)d_in[4];
  const float*     bk      = (const float*)d_in[5];
  const float*     Wv      = (const float*)d_in[6];
  const float*     bv      = (const float*)d_in[7];
  const float*     gamma   = (const float*)d_in[8];
  const float*     beta    = (const float*)d_in[9];

  char* ws = (char*)d_ws;
  bf16_t* featT = (bf16_t*)ws;                              // 8 MB
  bf16_t* wbf   = (bf16_t*)(ws + (8u << 20));               // 24 KB
  float*  gsum  = (float*)(ws + (8u << 20) + (32u << 10));  // 64 f32
  float*  gsq   = gsum + 64;
  float*  xbuf  = (float*)(ws + (8u << 20) + (64u << 10));  // 16 MB

  hipMemsetAsync(gsum, 0, 2 * 64 * sizeof(float), stream);
  k_prep_w<<<48, 256, 0, stream>>>(Wq, Wk, Wv, wbf);
  k_transpose<<<NBAT * (N_PTS / 64), 256, 0, stream>>>(feature, featT);
  k_main<<<(NBAT * N_PTS) / (WAVES * PTS_PER_WAVE), 128, 0, stream>>>(
      featT, knn, wbf, bq, bk, bv, xbuf);
  k_stats<<<256, 256, 0, stream>>>(xbuf, gsum, gsq);
  k_apply<<<NBAT * (N_PTS / 64), 256, 0, stream>>>(feature, xbuf, gsum, gsq, gamma,
                                                   beta, (float*)d_out);
}
